// ResBlock_27178553049603
// MI455X (gfx1250) — matrix operand order repaired and checked by reading, not compiled, not run on hardware
//
#include <hip/hip_runtime.h>
#include <hip/hip_bf16.h>

typedef __attribute__((ext_vector_type(16))) _Float16 v16h;
typedef __attribute__((ext_vector_type(2)))  _Float16 h2f16;
typedef __attribute__((ext_vector_type(8)))  float    v8f;
typedef __attribute__((ext_vector_type(4)))  unsigned u32x4;

#define COUT 128
#define WGVOX 128
#define OSTR 136   // padded f16 row stride for the output staging buffer

__device__ __forceinline__ unsigned short f2h(float f) {
    _Float16 h = (_Float16)f;
    return __builtin_bit_cast(unsigned short, h);
}
__device__ __forceinline__ float h2f(unsigned short u) {
    return (float)__builtin_bit_cast(_Float16, u);
}

union U4    { u32x4 q;    unsigned u[4]; };
union AFrag { u32x4 q[2]; v16h v; };
union HU    { unsigned u; h2f16 h; };

// ---------------- prep kernels ----------------

__global__ void k_init(float* stats, float* aId, float* bId) {
    int t = blockIdx.x * 256 + threadIdx.x;
    if (t < 1024) stats[t] = 0.f;
    if (t < 128) { aId[t] = 1.f; bId[t] = 0.f; }
}

__global__ void k_cvt(const float* __restrict__ src, unsigned short* __restrict__ dst, int total) {
    int i = blockIdx.x * 256 + threadIdx.x;
    if (i < total) dst[i] = f2h(src[i]);
}

// Repack W [9][CIN][128] f32 into WMMA B-fragment layout:
// [tap k][ktile t][coltile c][lane 0..31][elem 0..15] f16, where
// lane holds column N = c*16 + (lane&15), elem e -> K row = t*32 + (lane>>4)*8 + (e&7) + (e>>3)*16.
__global__ void k_prepw(const float* __restrict__ W, unsigned short* __restrict__ out,
                        int CIN, int total) {
    int i = blockIdx.x * 256 + threadIdx.x;
    if (i >= total) return;
    int e    = i & 15;
    int lane = (i >> 4) & 31;
    int c    = (i >> 9) & 7;
    int rest = i >> 12;
    int KT   = CIN >> 5;
    int t    = rest % KT;
    int k    = rest / KT;
    int col  = c * 16 + (lane & 15);
    int ci   = t * 32 + ((lane >> 4) << 3) + (e & 7) + ((e >> 3) << 4);
    out[i] = f2h(W[((size_t)k * CIN + ci) * COUT + col]);
}

// ---------------- fused gather-conv + lrelu + stats ----------------
// Yin:  [n, CIN] f16 (pre-BN y values); (a,b): per-channel BN affine folded from prev layer
// nbr:  [9, n] neighbor indices (idx==n => zero row)
// Wf:   fragment-layout f16 weights
// Yout: [n, 128] f16 = lrelu(conv) ; gsum/gsq: per-channel sum / sum-of-squares of Yout
// APPLY_BN=false: (a,b) is identity -> stage via GLOBAL_LOAD_ASYNC_TO_LDS (pure DMA gather)
template<int CIN, bool APPLY_BN>
__global__ void __launch_bounds__(256) k_conv(
    const unsigned short* __restrict__ Yin,
    const float* __restrict__ a, const float* __restrict__ b,
    const int* __restrict__ nbr,
    const unsigned short* __restrict__ Wf,
    unsigned short* __restrict__ Yout,
    float* __restrict__ gsum, float* __restrict__ gsq,
    int n)
{
    constexpr int KT   = CIN / 32;
    constexpr int CH   = CIN / 2;            // columns per staging thread (2 threads/row)
    constexpr int SROW = CIN + 8;            // padded f16 row stride (bank-conflict-free)
    __shared__ unsigned short sh[WGVOX * OSTR];                    // staging & output transpose
    __shared__ unsigned sAp[APPLY_BN ? CIN / 2 : 1];               // BN affine, packed f16 pairs
    __shared__ unsigned sBp[APPLY_BN ? CIN / 2 : 1];
    __shared__ float sstat[256];

    const int tid  = threadIdx.x;
    const int lane = tid & 31;
    const int w    = tid >> 5;               // wave id = output column tile
    const int base = blockIdx.x * WGVOX;

    sstat[tid] = 0.f;
    if (APPLY_BN && tid < CIN / 2) {
        HU pa, pb;
        pa.h = h2f16{(_Float16)a[2 * tid], (_Float16)a[2 * tid + 1]};
        pb.h = h2f16{(_Float16)b[2 * tid], (_Float16)b[2 * tid + 1]};
        sAp[tid] = pa.u;
        sBp[tid] = pb.u;
    }

    v8f acc[8];
    v8f vz = {};
#pragma unroll
    for (int m = 0; m < 8; ++m) acc[m] = vz;

    const int r    = tid >> 1;               // staging row 0..127
    const int half = tid & 1;
    const int vox  = base + r;
    const int hk   = lane >> 4;
    const int mrow = lane & 15;
    // workgroup-relative LDS byte offset of this thread's staging slot
    unsigned short* dstp = &sh[r * SROW + half * CH];
    const unsigned ldsoff = (unsigned)(size_t)dstp;

    for (int k = 0; k < 9; ++k) {
        __syncthreads();                     // protect sh + sAp/sBp
        // ---- gather 128 rows (sentinel -> zero) ----
        int idx = (vox < n) ? nbr[(size_t)k * n + vox] : n;
        if ((unsigned)idx < (unsigned)n) {
            if (!APPLY_BN) {
                // pure copy: async DMA global -> LDS, one b128 per 16B chunk.
                // INST_OFFSET advances LDS and global addresses together.
                unsigned long long ga =
                    (unsigned long long)(Yin + (size_t)idx * CIN + half * CH);
#pragma unroll
                for (int c8 = 0; c8 < CH / 8; ++c8) {
                    asm volatile("global_load_async_to_lds_b128 %0, %1, off offset:%2"
                                 :: "v"(ldsoff), "v"(ga), "i"(c8 * 16) : "memory");
                }
            } else {
                const u32x4* src =
                    reinterpret_cast<const u32x4*>(Yin + (size_t)idx * CIN + half * CH);
                u32x4* dst = reinterpret_cast<u32x4*>(dstp);
#pragma unroll
                for (int c8 = 0; c8 < CH / 8; ++c8) {
                    U4 in; in.q = src[c8];
                    U4 o;
                    int pb = half * (CH / 2) + c8 * 4;
#pragma unroll
                    for (int j = 0; j < 4; ++j) {
                        HU x;  x.u  = in.u[j];
                        HU aa; aa.u = sAp[pb + j];
                        HU bb; bb.u = sBp[pb + j];
                        HU rr; rr.h = x.h * aa.h + bb.h;   // v_pk_fma_f16
                        o.u[j] = rr.u;
                    }
                    dst[c8] = o.q;
                }
            }
        } else {
            u32x4 z = {0u, 0u, 0u, 0u};
            u32x4* dst = reinterpret_cast<u32x4*>(dstp);
#pragma unroll
            for (int c8 = 0; c8 < CH / 8; ++c8) dst[c8] = z;
        }
        if (!APPLY_BN) {
            asm volatile("s_wait_asynccnt 0x0" ::: "memory");
        }
        __syncthreads();

        // ---- WMMA accumulate: 8 voxel sub-tiles x KT k-tiles, A loads pipelined ----
#pragma unroll
        for (int t = 0; t < KT; ++t) {
            v16h bv = *reinterpret_cast<const v16h*>(
                Wf + ((((size_t)(k * KT + t) * 8 + w) * 32 + lane) << 4));
            // A 16x32 f16 layout: lanes 0-15 row M=lane, elems K0-7|K16-23;
            // lanes 16-31 row M=lane-16, elems K8-15|K24-31.
            AFrag cur;
            {
                const u32x4* sp = reinterpret_cast<const u32x4*>(
                    &sh[mrow * SROW + 32 * t + 8 * hk]);
                cur.q[0] = sp[0];
                cur.q[1] = sp[2];
            }
#pragma unroll
            for (int m = 0; m < 8; ++m) {
                AFrag nxt = cur;
                if (m < 7) {
                    const u32x4* sp = reinterpret_cast<const u32x4*>(
                        &sh[(16 * (m + 1) + mrow) * SROW + 32 * t + 8 * hk]);
                    nxt.q[0] = sp[0];
                    nxt.q[1] = sp[2];
                }
                acc[m] = __builtin_amdgcn_wmma_f32_16x16x32_f16(
                    false, cur.v, false, bv, (short)0, acc[m], false, false);
                cur = nxt;
            }
        }
    }

    // ---- epilogue: lrelu + stats, transpose through LDS, coalesced b128 stores ----
    __syncthreads();
    const int ch = w * 16 + (lane & 15);
    float lsum = 0.f, lsq = 0.f;
#pragma unroll
    for (int m = 0; m < 8; ++m) {
#pragma unroll
        for (int j = 0; j < 8; ++j) {
            int row = 16 * m + j + 8 * hk;      // C/D: VGPR j -> row j + 8*(lane/16)
            float v = acc[m][j];
            float y = v >= 0.f ? v : 0.01f * v;
            sh[row * OSTR + ch] = f2h(y);
            if (base + row < n) { lsum += y; lsq += y * y; }
        }
    }
    atomicAdd(&sstat[ch], lsum);
    atomicAdd(&sstat[128 + ch], lsq);
    __syncthreads();
    {
        int vv = base + r;
        if (vv < n) {
            const u32x4* s4 = reinterpret_cast<const u32x4*>(&sh[r * OSTR + half * 64]);
            u32x4* d4 = reinterpret_cast<u32x4*>(Yout + (size_t)vv * COUT + half * 64);
#pragma unroll
            for (int q = 0; q < 8; ++q) d4[q] = s4[q];
        }
    }
    if (tid < 128) {
        atomicAdd(&gsum[tid], sstat[tid]);
        atomicAdd(&gsq[tid],  sstat[128 + tid]);
    }
}

// ---------------- BN stat folding: (sum,sumsq,g,b) -> per-channel (scale,bias) ----------------
__global__ void k_bnscale(const float* __restrict__ ssum, const float* __restrict__ ssq,
                          const float* __restrict__ g, const float* __restrict__ b,
                          float* __restrict__ ao, float* __restrict__ bo, float invN) {
    int c = threadIdx.x;
    float m = ssum[c] * invN;
    float v = ssq[c] * invN - m * m;
    float s = g[c] * rsqrtf(v + 1e-5f);
    ao[c] = s;
    bo[c] = b[c] - m * s;
}

// ---------------- final: out = bn(y_s) + bn(y_r) ----------------
__global__ void k_final(const unsigned short* __restrict__ Ys, const float* __restrict__ aS,
                        const float* __restrict__ bS,
                        const unsigned short* __restrict__ Yr, const float* __restrict__ aR,
                        const float* __restrict__ bR,
                        float* __restrict__ out, int total) {
    int i = blockIdx.x * 256 + threadIdx.x;
    if (i >= total) return;
    int c = i & 127;
    out[i] = (h2f(Ys[i]) * aS[c] + bS[c]) + (h2f(Yr[i]) * aR[c] + bR[c]);
}

extern "C" void kernel_launch(void* const* d_in, const int* in_sizes, int n_in,
                              void* d_out, int out_size, void* d_ws, size_t ws_size,
                              hipStream_t stream) {
    (void)n_in; (void)out_size; (void)ws_size;
    const float* features = (const float*)d_in[0];
    const int*   nbr31    = (const int*)d_in[1];
    const int*   nbr13    = (const int*)d_in[2];
    const float* Wc1  = (const float*)d_in[3];
    const float* Wc12 = (const float*)d_in[4];
    const float* Wc2  = (const float*)d_in[5];
    const float* Wc3  = (const float*)d_in[6];
    const float* g0  = (const float*)d_in[7];  const float* b0  = (const float*)d_in[8];
    const float* g02 = (const float*)d_in[9];  const float* b02 = (const float*)d_in[10];
    const float* g1  = (const float*)d_in[11]; const float* b1  = (const float*)d_in[12];
    const float* g2  = (const float*)d_in[13]; const float* b2  = (const float*)d_in[14];

    const int n = in_sizes[0] / 64;

    char* p = (char*)d_ws;
    auto carve = [&](size_t bytes) -> char* {
        char* r = p; p += (bytes + 255) & ~(size_t)255; return r;
    };
    unsigned short* featH = (unsigned short*)carve((size_t)n * 64 * 2);
    unsigned short* X     = (unsigned short*)carve((size_t)n * 128 * 2);   // y1 then reused for y_r1
    unsigned short* Ys2   = (unsigned short*)carve((size_t)n * 128 * 2);
    unsigned short* Yr2   = (unsigned short*)carve((size_t)n * 128 * 2);
    unsigned short* Wf1   = (unsigned short*)carve((size_t)9 * 2 * 8 * 512 * 2);
    unsigned short* Wf12  = (unsigned short*)carve((size_t)9 * 4 * 8 * 512 * 2);
    unsigned short* Wf2   = (unsigned short*)carve((size_t)9 * 2 * 8 * 512 * 2);
    unsigned short* Wf3   = (unsigned short*)carve((size_t)9 * 4 * 8 * 512 * 2);
    float* stats = (float*)carve(1024 * 4);   // 4 convs x (sum[128], sumsq[128])
    float* ab    = (float*)carve(1280 * 4);   // 4 convs x (a[128], b[128]) + identity a/b
    float* aId = ab + 1024;
    float* bId = ab + 1152;

    k_init<<<4, 256, 0, stream>>>(stats, aId, bId);
    { int tot = n * 64;          k_cvt  <<<(tot + 255) / 256, 256, 0, stream>>>(features, featH, tot); }
    { int tot = 9 * 2 * 8 * 512; k_prepw<<<(tot + 255) / 256, 256, 0, stream>>>(Wc1,  Wf1,  64,  tot); }
    { int tot = 9 * 4 * 8 * 512; k_prepw<<<(tot + 255) / 256, 256, 0, stream>>>(Wc12, Wf12, 128, tot); }
    { int tot = 9 * 2 * 8 * 512; k_prepw<<<(tot + 255) / 256, 256, 0, stream>>>(Wc2,  Wf2,  64,  tot); }
    { int tot = 9 * 4 * 8 * 512; k_prepw<<<(tot + 255) / 256, 256, 0, stream>>>(Wc3,  Wf3,  128, tot); }

    const int   nwg  = (n + WGVOX - 1) / WGVOX;
    const float invN = 1.0f / (float)n;

    // shortcut: conv(3,1,3) -> lrelu -> bn -> conv(1,3,3) -> lrelu -> bn
    k_conv<64, false> <<<nwg, 256, 0, stream>>>(featH, aId,      bId,      nbr31, Wf1,  X,   stats + 0,   stats + 128, n);
    k_bnscale         <<<1, 128, 0, stream>>>(stats + 0,   stats + 128, g0,  b0,  ab + 0,   ab + 128, invN);
    k_conv<128, true> <<<nwg, 256, 0, stream>>>(X,     ab + 0,   ab + 128, nbr13, Wf12, Ys2, stats + 256, stats + 384, n);
    k_bnscale         <<<1, 128, 0, stream>>>(stats + 256, stats + 384, g02, b02, ab + 256, ab + 384, invN);
    // main: conv(1,3,3) -> lrelu -> bn -> conv(3,1,3) -> lrelu -> bn
    k_conv<64, false> <<<nwg, 256, 0, stream>>>(featH, aId,      bId,      nbr13, Wf2,  X,   stats + 512, stats + 640, n);
    k_bnscale         <<<1, 128, 0, stream>>>(stats + 512, stats + 640, g1,  b1,  ab + 512, ab + 640, invN);
    k_conv<128, true> <<<nwg, 256, 0, stream>>>(X,     ab + 512, ab + 640, nbr31, Wf3,  Yr2, stats + 768, stats + 896, n);
    k_bnscale         <<<1, 128, 0, stream>>>(stats + 768, stats + 896, g2,  b2,  ab + 768, ab + 896, invN);

    { int tot = n * 128;
      k_final<<<(tot + 255) / 256, 256, 0, stream>>>(Ys2, ab + 256, ab + 384,
                                                     Yr2, ab + 768, ab + 896,
                                                     (float*)d_out, tot); }
}
